// GAT_19825569038771
// MI455X (gfx1250) — compile-verified, matrix operand order checked by reading, not run on hardware
//
#include <hip/hip_runtime.h>
#include <hip/hip_bf16.h>
#include <math.h>

#define N_NODES   50000
#define N_GRAPHS  500
#define NPG       100
#define N_EDGES_C 500000
#define F_IN      16
#define F_GRAPH   4
#define HID       64
#define NEG_SLOPE 0.2f
#define EPS_C     1e-5f

typedef __attribute__((ext_vector_type(16))) __bf16    v16bf;
typedef __attribute__((ext_vector_type(8)))  float     v8f;
typedef __attribute__((ext_vector_type(4)))  unsigned  u32x4;

union fragU { v16bf v; u32x4 q[2]; };

// ---------------- f32 -> bf16 converters (zero-padded K) ----------------
__global__ void cvt_bf16_rows(const float* __restrict__ src, __bf16* __restrict__ dst,
                              int rows, int cols, int ldo) {
  long t = (long)blockIdx.x * blockDim.x + threadIdx.x;
  long total = (long)rows * ldo;
  if (t >= total) return;
  int r = (int)(t / ldo), c = (int)(t % ldo);
  float v = (c < cols) ? src[(long)r * cols + c] : 0.f;
  dst[t] = (__bf16)v;
}

// W[K,N] f32 -> WT[N, ldo] bf16 (transposed, K zero-padded to ldo)
__global__ void cvt_bf16_T(const float* __restrict__ src, __bf16* __restrict__ dst,
                           int K, int N, int ldo) {
  long t = (long)blockIdx.x * blockDim.x + threadIdx.x;
  long total = (long)N * ldo;
  if (t >= total) return;
  int n = (int)(t / ldo), k = (int)(t % ldo);
  float v = (k < K) ? src[(long)k * N + n] : 0.f;
  dst[t] = (__bf16)v;
}

// ---------------- WMMA bf16 GEMM:  C[M,N] = A[M,Kp] * BT[N,Kp]^T ----------------
// block = 256 threads (8 waves). Wave w: rows [bx*128+16w, +16), cols [by*64, +64)
// (4 accumulators). Requires: M % 16 == 0, Kp % 32 == 0, N % 64 == 0.
// Fragment layout per ISA (CDNA5 ISA §7.12.2, 16-bit 16x32):
//   lanes 0-15 : elems 0..7 -> K=k0+0..7,  elems 8..15 -> K=k0+16..23
//   lanes 16-31: elems 0..7 -> K=k0+8..15, elems 8..15 -> K=k0+24..31
// -> per lane: two contiguous 8-element (16B) chunks at k0+base and k0+base+16.
__global__ void gemm_wmma_bf16(const __bf16* __restrict__ A,
                               const __bf16* __restrict__ BT,
                               float* __restrict__ C,
                               int M, int Kp, int N) {
  const int lane = threadIdx.x & 31;
  const int wave = threadIdx.x >> 5;
  const int m0   = blockIdx.x * 128 + wave * 16;
  if (m0 >= M) return;                       // wave-uniform tail exit (M % 16 == 0)
  const int n0   = blockIdx.y * 64;
  const int l15  = lane & 15;
  const int kbase = (lane >> 4) << 3;        // 0 or 8

  const __bf16* arow = A  + (size_t)(m0 + l15) * Kp + kbase;
  const __bf16* b0   = BT + (size_t)(n0 + l15)      * Kp + kbase;
  const __bf16* b1   = b0 + (size_t)16 * Kp;
  const __bf16* b2   = b0 + (size_t)32 * Kp;
  const __bf16* b3   = b0 + (size_t)48 * Kp;

  v8f acc0 = {}, acc1 = {}, acc2 = {}, acc3 = {};
  for (int k0 = 0; k0 < Kp; k0 += 32) {
    if (k0 + 32 < Kp)
      __builtin_prefetch(arow + k0 + 32, 0, 3);   // WGP-scope: pull into all levels
    fragU fa, fb0, fb1, fb2, fb3;
    const u32x4* pa = (const u32x4*)(arow + k0);
    fa.q[0] = pa[0]; fa.q[1] = pa[2];             // +0 / +16 elements
    const u32x4* p0 = (const u32x4*)(b0 + k0);
    fb0.q[0] = p0[0]; fb0.q[1] = p0[2];
    const u32x4* p1 = (const u32x4*)(b1 + k0);
    fb1.q[0] = p1[0]; fb1.q[1] = p1[2];
    const u32x4* p2 = (const u32x4*)(b2 + k0);
    fb2.q[0] = p2[0]; fb2.q[1] = p2[2];
    const u32x4* p3 = (const u32x4*)(b3 + k0);
    fb3.q[0] = p3[0]; fb3.q[1] = p3[2];
    acc0 = __builtin_amdgcn_wmma_f32_16x16x32_bf16(false, fa.v, false, fb0.v, (short)0, acc0, false, false);
    acc1 = __builtin_amdgcn_wmma_f32_16x16x32_bf16(false, fa.v, false, fb1.v, (short)0, acc1, false, false);
    acc2 = __builtin_amdgcn_wmma_f32_16x16x32_bf16(false, fa.v, false, fb2.v, (short)0, acc2, false, false);
    acc3 = __builtin_amdgcn_wmma_f32_16x16x32_bf16(false, fa.v, false, fb3.v, (short)0, acc3, false, false);
  }
  // D layout: VGPR v -> row m0+v (lanes 0-15) / m0+8+v (lanes 16-31), col = n0' + (lane&15)
  const int r0 = m0 + kbase;                 // kbase == 8*(lane>=16)
  float* crow = C + (size_t)r0 * N + n0 + l15;
#pragma unroll
  for (int v = 0; v < 8; ++v) {
    float* cr = crow + (size_t)v * N;
    cr[0]  = acc0[v];
    cr[16] = acc1[v];
    cr[32] = acc2[v];
    cr[48] = acc3[v];
  }
}

// ---------------- GraphNorm: one block (128 thr) per graph ----------------
__global__ void graph_norm_kernel(const float* __restrict__ x,
                                  const float* __restrict__ gw,
                                  const float* __restrict__ gb,
                                  const float* __restrict__ gms,
                                  float* __restrict__ xn) {
  __shared__ float part[128];
  __shared__ float meanv[F_IN];
  __shared__ float varv[F_IN];
  const int g = blockIdx.x;
  const int t = threadIdx.x;
  const int f = t & 15;
  const int c = t >> 4;          // 0..7
  const float* xg = x + (long)g * NPG * F_IN;
  float s = 0.f;
  for (int n = c; n < NPG; n += 8) s += xg[n * F_IN + f];
  part[t] = s;
  __syncthreads();
  if (t < F_IN) {
    float m = 0.f;
    for (int cc = 0; cc < 8; ++cc) m += part[cc * 16 + t];
    meanv[t] = m / (float)NPG;
  }
  __syncthreads();
  const float mm = meanv[f] * gms[f];
  s = 0.f;
  for (int n = c; n < NPG; n += 8) { float v = xg[n * F_IN + f] - mm; s += v * v; }
  part[t] = s;
  __syncthreads();
  if (t < F_IN) {
    float v = 0.f;
    for (int cc = 0; cc < 8; ++cc) v += part[cc * 16 + t];
    varv[t] = v / (float)NPG;
  }
  __syncthreads();
  const float inv = rsqrtf(varv[f] + EPS_C) * gw[f];
  const float bb  = gb[f];
  float* og = xn + (long)g * NPG * F_IN;
  for (int n = c; n < NPG; n += 8)
    og[n * F_IN + f] = (xg[n * F_IN + f] - mm) * inv + bb;
}

// ---------------- attention scalar scores per (node, head) ----------------
__global__ void attn_scores_kernel(const float* __restrict__ P,
                                   const float* __restrict__ att_s,
                                   const float* __restrict__ att_d,
                                   float* __restrict__ a_src,
                                   float* __restrict__ a_dst, int H) {
  int t = blockIdx.x * blockDim.x + threadIdx.x;
  if (t >= N_NODES * H) return;
  int n = t / H, h = t - n * H;
  const float* hp = P + (long)n * H * HID + h * HID;
  const float* as = att_s + h * HID;
  const float* ad = att_d + h * HID;
  float ss = 0.f, sd = 0.f;
  for (int c = 0; c < HID; ++c) { float v = hp[c]; ss += v * as[c]; sd += v * ad[c]; }
  a_src[t] = ss;
  a_dst[t] = sd;
}

// order-preserving float<->uint encoding for atomic segment max
__device__ __forceinline__ unsigned f2ord(float x) {
  unsigned b = __float_as_uint(x);
  return (b & 0x80000000u) ? ~b : (b | 0x80000000u);
}
__device__ __forceinline__ float ord2f(unsigned u) {
  unsigned b = (u & 0x80000000u) ? (u ^ 0x80000000u) : ~u;
  return __uint_as_float(b);
}

__device__ __forceinline__ void edge_sd(int e, const int* esrc, const int* edst,
                                        int& s, int& d) {
  if (e < N_EDGES_C) { s = esrc[e]; d = edst[e]; }
  else               { s = e - N_EDGES_C; d = s; }   // implicit self-loops
}

__global__ void edge_max_kernel(const int* __restrict__ esrc, const int* __restrict__ edst,
                                const float* __restrict__ a_src, const float* __restrict__ a_dst,
                                unsigned* __restrict__ maxu, int H, int Etot) {
  int t = blockIdx.x * blockDim.x + threadIdx.x;
  if (t >= Etot * H) return;
  int e = t / H, h = t - e * H, s, d;
  edge_sd(e, esrc, edst, s, d);
  float ev = a_src[s * H + h] + a_dst[d * H + h];
  ev = (ev > 0.f) ? ev : ev * NEG_SLOPE;
  atomicMax(&maxu[d * H + h], f2ord(ev));
}

__global__ void edge_expsum_kernel(const int* __restrict__ esrc, const int* __restrict__ edst,
                                   const float* __restrict__ a_src, const float* __restrict__ a_dst,
                                   const unsigned* __restrict__ maxu,
                                   float* __restrict__ sumf, int H, int Etot) {
  int t = blockIdx.x * blockDim.x + threadIdx.x;
  if (t >= Etot * H) return;
  int e = t / H, h = t - e * H, s, d;
  edge_sd(e, esrc, edst, s, d);
  float ev = a_src[s * H + h] + a_dst[d * H + h];
  ev = (ev > 0.f) ? ev : ev * NEG_SLOPE;
  atomicAdd(&sumf[d * H + h], expf(ev - ord2f(maxu[d * H + h])));
}

// one block per edge; blockDim = H*HID lanes scatter features
__global__ void edge_aggregate_kernel(const int* __restrict__ esrc, const int* __restrict__ edst,
                                      const float* __restrict__ a_src, const float* __restrict__ a_dst,
                                      const unsigned* __restrict__ maxu, const float* __restrict__ sumf,
                                      const float* __restrict__ P, float* __restrict__ out,
                                      int H, int Etot) {
  int e = blockIdx.x;
  int c = threadIdx.x;
  int h = c / HID, s, d;
  edge_sd(e, esrc, edst, s, d);
  float ev = a_src[s * H + h] + a_dst[d * H + h];
  ev = (ev > 0.f) ? ev : ev * NEG_SLOPE;
  float alpha = expf(ev - ord2f(maxu[d * H + h])) / sumf[d * H + h];
  atomicAdd(&out[(long)d * H * HID + c], P[(long)s * H * HID + c] * alpha);
}

__global__ void bias_act_kernel(float* __restrict__ out, const float* __restrict__ bias,
                                long total, int HC, int act) {
  long t = (long)blockIdx.x * blockDim.x + threadIdx.x;
  if (t >= total) return;
  int c = (int)(t % HC);
  float v = out[t] + bias[c];
  if (act) v = (v > 0.f) ? v : expm1f(v);   // ELU
  out[t] = v;
}

__device__ __forceinline__ float selu_f(float x) {
  const float sc = 1.0507009873554805f, al = 1.6732632423543772f;
  return (x > 0.f) ? sc * x : sc * al * expm1f(x);
}

// pool + concat + BN(eval) + 2x SELU dense + logits + softmax; block(HID) per graph
__global__ void head_kernel(const float* __restrict__ Q, const float* __restrict__ ginp,
                            const float* __restrict__ bn_g, const float* __restrict__ bn_b,
                            const float* __restrict__ bn_m, const float* __restrict__ bn_v,
                            const float* __restrict__ Wd1, const float* __restrict__ bd1,
                            const float* __restrict__ Wd2, const float* __restrict__ bd2,
                            const float* __restrict__ Wo,  const float* __restrict__ bo,
                            float* __restrict__ outp) {
  __shared__ float feat[HID + F_GRAPH];
  __shared__ float t1[HID];
  __shared__ float t2[HID];
  __shared__ float logit[2];
  const int g = blockIdx.x, j = threadIdx.x;
  const float* qg = Q + (long)g * NPG * HID;
  float s = 0.f;
  for (int n = 0; n < NPG; ++n) s += qg[n * HID + j];
  float raw  = s / (float)NPG;
  float raw2 = (j < F_GRAPH) ? ginp[g * F_GRAPH + j] : 0.f;
  feat[j] = (raw - bn_m[j]) * rsqrtf(bn_v[j] + EPS_C) * bn_g[j] + bn_b[j];
  if (j < F_GRAPH) {
    int jj = HID + j;
    feat[jj] = (raw2 - bn_m[jj]) * rsqrtf(bn_v[jj] + EPS_C) * bn_g[jj] + bn_b[jj];
  }
  __syncthreads();
  s = bd1[j];
  for (int i = 0; i < HID + F_GRAPH; ++i) s += feat[i] * Wd1[i * HID + j];
  t1[j] = selu_f(s);
  __syncthreads();
  s = bd2[j];
  for (int i = 0; i < HID; ++i) s += t1[i] * Wd2[i * HID + j];
  t2[j] = selu_f(s);
  __syncthreads();
  if (j < 2) {
    s = bo[j];
    for (int i = 0; i < HID; ++i) s += t2[i] * Wo[i * 2 + j];
    logit[j] = s;
  }
  __syncthreads();
  if (j < 2) {
    float m  = fmaxf(logit[0], logit[1]);
    float e0 = expf(logit[0] - m), e1 = expf(logit[1] - m);
    outp[g * 2 + j] = expf(logit[j] - m) / (e0 + e1);
  }
}

extern "C" void kernel_launch(void* const* d_in, const int* in_sizes, int n_in,
                              void* d_out, int out_size, void* d_ws, size_t ws_size,
                              hipStream_t stream) {
  const float* x    = (const float*)d_in[0];
  const int*   eidx = (const int*)d_in[1];
  const float* ginp = (const float*)d_in[2];
  const float* gn_w = (const float*)d_in[4];
  const float* gn_b = (const float*)d_in[5];
  const float* gn_ms= (const float*)d_in[6];
  const float* W1   = (const float*)d_in[7];
  const float* as1  = (const float*)d_in[8];
  const float* ad1  = (const float*)d_in[9];
  const float* b1   = (const float*)d_in[10];
  const float* W2   = (const float*)d_in[11];
  const float* as2  = (const float*)d_in[12];
  const float* ad2  = (const float*)d_in[13];
  const float* b2   = (const float*)d_in[14];
  const float* W3   = (const float*)d_in[15];
  const float* as3  = (const float*)d_in[16];
  const float* ad3  = (const float*)d_in[17];
  const float* b3   = (const float*)d_in[18];
  const float* bn_g = (const float*)d_in[19];
  const float* bn_b = (const float*)d_in[20];
  const float* bn_m = (const float*)d_in[21];
  const float* bn_v = (const float*)d_in[22];
  const float* Wd1  = (const float*)d_in[23];
  const float* bd1  = (const float*)d_in[24];
  const float* Wd2  = (const float*)d_in[25];
  const float* bd2  = (const float*)d_in[26];
  const float* Wo   = (const float*)d_in[27];
  const float* bo   = (const float*)d_in[28];

  const int E = in_sizes[1] / 2;
  const int* esrc = eidx;
  const int* edst = eidx + E;
  const int Etot  = E + N_NODES;

  // workspace layout (all offsets multiples of 256B)
  char* ws = (char*)d_ws;
  size_t off = 0;
  float*    XN   = (float*)(ws + off); off += (size_t)N_NODES * F_IN * 4;        //  3.2 MB
  float*    P    = (float*)(ws + off); off += (size_t)N_NODES * 256 * 4;         // 51.2 MB
  float*    Q    = (float*)(ws + off); off += (size_t)N_NODES * 256 * 4;         // 51.2 MB
  __bf16*   Abf  = (__bf16*)(ws + off); off += (size_t)N_NODES * 256 * 2;        // 25.6 MB
  __bf16*   BTbf = (__bf16*)(ws + off); off += (size_t)256 * 256 * 2;            // 128 KB
  float*    ASRC = (float*)(ws + off); off += (size_t)N_NODES * 4 * 4;
  float*    ADST = (float*)(ws + off); off += (size_t)N_NODES * 4 * 4;
  unsigned* MAXU = (unsigned*)(ws + off); off += (size_t)N_NODES * 4 * 4;
  float*    SUMF = (float*)(ws + off); off += (size_t)N_NODES * 4 * 4;

  auto run_gemm = [&](const float* Ain, int K, const float* W, int N) {
    const int Kp = (K + 31) & ~31;                 // zero-padded K
    long tA = (long)N_NODES * Kp;
    cvt_bf16_rows<<<(int)((tA + 255) / 256), 256, 0, stream>>>(Ain, Abf, N_NODES, K, Kp);
    long tB = (long)N * Kp;
    cvt_bf16_T<<<(int)((tB + 255) / 256), 256, 0, stream>>>(W, BTbf, K, N, Kp);
    dim3 grid((N_NODES + 127) / 128, N / 64);
    gemm_wmma_bf16<<<grid, 256, 0, stream>>>(Abf, BTbf, P, N_NODES, Kp, N);
  };

  auto run_attn = [&](const float* Pin, float* Qout, const float* att_s, const float* att_d,
                      const float* bias, int H, int act) {
    const int HC = H * HID;
    attn_scores_kernel<<<(N_NODES * H + 255) / 256, 256, 0, stream>>>(Pin, att_s, att_d, ASRC, ADST, H);
    hipMemsetAsync(MAXU, 0, (size_t)N_NODES * H * sizeof(unsigned), stream);
    hipMemsetAsync(SUMF, 0, (size_t)N_NODES * H * sizeof(float), stream);
    hipMemsetAsync(Qout, 0, (size_t)N_NODES * HC * sizeof(float), stream);
    edge_max_kernel<<<(Etot * H + 255) / 256, 256, 0, stream>>>(esrc, edst, ASRC, ADST, MAXU, H, Etot);
    edge_expsum_kernel<<<(Etot * H + 255) / 256, 256, 0, stream>>>(esrc, edst, ASRC, ADST, MAXU, SUMF, H, Etot);
    edge_aggregate_kernel<<<Etot, HC, 0, stream>>>(esrc, edst, ASRC, ADST, MAXU, SUMF, Pin, Qout, H, Etot);
    bias_act_kernel<<<(int)(((long)N_NODES * HC + 255) / 256), 256, 0, stream>>>(
        Qout, bias, (long)N_NODES * HC, HC, act);
  };

  // 1) GraphNorm
  graph_norm_kernel<<<N_GRAPHS, 128, 0, stream>>>(x, gn_w, gn_b, gn_ms, XN);

  // 2) GAT layer 1: 16 -> 256
  run_gemm(XN, F_IN, W1, 4 * HID);
  run_attn(P, Q, as1, ad1, b1, 4, 1);
  // 3) GAT layer 2: 256 -> 256
  run_gemm(Q, 4 * HID, W2, 4 * HID);
  run_attn(P, Q, as2, ad2, b2, 4, 1);
  // 4) GAT layer 3: 256 -> 64 (single head, no activation)
  run_gemm(Q, 4 * HID, W3, HID);
  run_attn(P, Q, as3, ad3, b3, 1, 0);
  // 5) pool + BN + dense head + softmax
  head_kernel<<<N_GRAPHS, HID, 0, stream>>>(Q, ginp, bn_g, bn_b, bn_m, bn_v,
                                            Wd1, bd1, Wd2, bd2, Wo, bo, (float*)d_out);
}
